// ContentAdaptiveSparsity_71305047048516
// MI455X (gfx1250) — matrix operand order repaired and checked by reading, not compiled, not run on hardware
//
#include <hip/hip_runtime.h>
#include <hip/hip_bf16.h>

typedef __attribute__((ext_vector_type(16))) _Float16 v16h;
typedef __attribute__((ext_vector_type(8)))  float    v8f;
typedef __attribute__((ext_vector_type(4)))  float    v4f;   // native vector for NT loads

#define B_   4
#define S_   8192
#define H_   16
#define D_   128
#define NB   64
#define HID  32
#define HID2 16
#define ROWS (B_*NB*H_)   // 4096 pooled rows per side

// ---------------------------------------------------------------------------
// Kernel 1: block-mean pooling.  Streams 512MB of f32 (the roofline term).
// grid = 1024 blocks x 256 thr.  Per (src,b,i) tile two column-split blocks,
// each thread owns one float4 column group, 128-row reduction, NT b128 loads.
// Output stored as f16 (only consumed by WMMA GEMMs downstream).
// ---------------------------------------------------------------------------
__global__ void pool_kernel(const float* __restrict__ q, const float* __restrict__ k,
                            _Float16* __restrict__ qavg, _Float16* __restrict__ kavg) {
    int id = blockIdx.x;
    int srcsel = id >> 9;          // 512 blocks per source
    int r  = id & 511;
    int bi = r >> 1, cs = r & 1;   // column-split half (1024 cols each)
    int b  = bi >> 6, i = bi & 63;
    const float* src = srcsel ? k : q;
    _Float16*    dst = srcsel ? kavg : qavg;
    size_t base = ((size_t)b * S_ + (size_t)i * 128) * (H_ * D_) + (size_t)cs * 1024;
    int tid = threadIdx.x;

    float a0 = 0.f, a1 = 0.f, a2 = 0.f, a3 = 0.f;
    for (int s = 0; s < 128; ++s) {
        const v4f* p = (const v4f*)(src + base + (size_t)s * (H_ * D_)) + tid;
        v4f v = __builtin_nontemporal_load(p);
        a0 += v.x; a1 += v.y; a2 += v.z; a3 += v.w;
    }
    const float invn = 1.0f / 128.0f;
    float vals[4] = {a0 * invn, a1 * invn, a2 * invn, a3 * invn};
    #pragma unroll
    for (int c = 0; c < 4; ++c) {
        int col = cs * 1024 + tid * 4 + c;
        int h = col >> 7, d = col & 127;
        int row = (b * NB + i) * H_ + h;
        dst[(size_t)row * D_ + d] = (_Float16)vals[c];
    }
}

// ---------------------------------------------------------------------------
// Kernel 2: convert the two GEMM weight matrices to f16 once.
// ---------------------------------------------------------------------------
__global__ void prep_w_kernel(const float* __restrict__ w_imp1, const float* __restrict__ w_int1,
                              _Float16* __restrict__ w1h, _Float16* __restrict__ wi1h) {
    int t = blockIdx.x * blockDim.x + threadIdx.x;
    if (t < D_ * HID)      w1h[t]  = (_Float16)w_imp1[t];
    if (t < 2 * D_ * HID)  wi1h[t] = (_Float16)w_int1[t];
}

// ---------------------------------------------------------------------------
// Kernel 3: fused WMMA MLP.  One wave per 16-row tile of X=[4096,128] (f16).
//   h1   = relu(X @ w_imp1 + b_imp1)           (WMMA, N=32 -> 2 tiles, K=128 -> 4 steps)
//   part = X @ w_int1[src*128 : src*128+128]   (WMMA, same shape, no bias yet)
//   imp  = sigmoid(relu(h1 @ w_imp2 + b2) @ w_imp3 + b3)   (scalar tail, 16 lanes)
// Fragment packing per CDNA5 ISA 7.12.2 (wave32).
// ---------------------------------------------------------------------------
__global__ void __launch_bounds__(32) mlp_wmma_kernel(
    const _Float16* __restrict__ qavg, const _Float16* __restrict__ kavg,
    const _Float16* __restrict__ w1h,  const _Float16* __restrict__ wi1h,
    const float* __restrict__ b1, const float* __restrict__ w2,
    const float* __restrict__ b2, const float* __restrict__ w3,
    const float* __restrict__ b3,
    float* __restrict__ qpart, float* __restrict__ kpart,
    float* __restrict__ qimp,  float* __restrict__ kimp) {
    __shared__ _Float16 xt[16 * D_];        // 4KB  : X tile
    __shared__ _Float16 wl[2 * D_ * HID];   // 16KB : [0..4095]=w_imp1, [4096..]=w_int1 half
    __shared__ float    h1s[16][HID];       // 2KB

    int id = blockIdx.x;
    int srcsel = id >> 8;                   // 256 tiles per source
    int tile = id & 255;
    int row0 = tile * 16;
    const _Float16* X  = srcsel ? kavg : qavg;
    const _Float16* wp = wi1h + srcsel * (D_ * HID);
    float* part = srcsel ? kpart : qpart;
    float* imp  = srcsel ? kimp  : qimp;
    int lane = threadIdx.x;

    // stage tile + weights (16 consecutive rows of X are contiguous)
    const uint4* gx = (const uint4*)(X + (size_t)row0 * D_);
    uint4* lx = (uint4*)xt;
    for (int t = lane; t < 256; t += 32) lx[t] = gx[t];
    const uint4* gw1 = (const uint4*)w1h;
    const uint4* gwp = (const uint4*)wp;
    uint4* lw = (uint4*)wl;
    for (int t = lane; t < 1024; t += 32) lw[t] = (t < 512) ? gw1[t] : gwp[t - 512];
    __syncthreads();

    int m = lane & 15, half = lane >> 4;
    v8f acc_h1a = {}, acc_h1b = {}, acc_pa = {}, acc_pb = {};
    for (int kb = 0; kb < 4; ++kb) {
        // A fragment: 16x32 f16, lane m=lane&15 (+half), VGPR pair K layout
        v16h a;
        #pragma unroll
        for (int v = 0; v < 8; ++v) {
            int kk = ((v < 4) ? 2 * v : 8 + 2 * v) + 8 * half + kb * 32;
            a[2 * v]     = xt[m * D_ + kk];
            a[2 * v + 1] = xt[m * D_ + kk + 1];
        }
        // B fragments: 32x16 f16, lane = N column (lane&15), 16 K values per lane in same K order as A
        int krow = kb * 32 + 8 * half;
        v16h bf0, bf1, pf0, pf1;
        #pragma unroll
        for (int e = 0; e < 16; ++e) {
            int kr = krow + ((e < 8) ? e : 8 + e);
            bf0[e] = wl[kr * HID + m];
            bf1[e] = wl[kr * HID + 16 + m];
            pf0[e] = wl[4096 + kr * HID + m];
            pf1[e] = wl[4096 + kr * HID + 16 + m];
        }
        acc_h1a = __builtin_amdgcn_wmma_f32_16x16x32_f16(false, a, false, bf0, (short)0, acc_h1a, false, false);
        acc_h1b = __builtin_amdgcn_wmma_f32_16x16x32_f16(false, a, false, bf1, (short)0, acc_h1b, false, false);
        acc_pa  = __builtin_amdgcn_wmma_f32_16x16x32_f16(false, a, false, pf0, (short)0, acc_pa,  false, false);
        acc_pb  = __builtin_amdgcn_wmma_f32_16x16x32_f16(false, a, false, pf1, (short)0, acc_pb,  false, false);
    }
    // D layout: VGPR r -> M = r + 8*half, N = lane&15
    int n = lane & 15;
    #pragma unroll
    for (int r = 0; r < 8; ++r) {
        int mm = r + 8 * half;
        h1s[mm][n]      = fmaxf(acc_h1a[r] + b1[n],      0.0f);
        h1s[mm][16 + n] = fmaxf(acc_h1b[r] + b1[16 + n], 0.0f);
        part[(size_t)(row0 + mm) * HID + n]      = acc_pa[r];
        part[(size_t)(row0 + mm) * HID + 16 + n] = acc_pb[r];
    }
    __syncthreads();
    if (lane < 16) {
        int mm = lane;
        float h2r[HID2];
        #pragma unroll
        for (int j2 = 0; j2 < HID2; ++j2) {
            float s = b2[j2];
            for (int c = 0; c < HID; ++c) s += h1s[mm][c] * w2[c * HID2 + j2];
            h2r[j2] = fmaxf(s, 0.0f);
        }
        float s3 = b3[0];
        #pragma unroll
        for (int j2 = 0; j2 < HID2; ++j2) s3 += h2r[j2] * w3[j2];
        imp[row0 + mm] = 1.0f / (1.0f + __expf(-s3));
    }
}

// ---------------------------------------------------------------------------
// Kernel 4: pairwise interaction + combine.  262144 outputs, L2-resident data.
// combined[b][i][j][h] == flat index order used below.
// ---------------------------------------------------------------------------
__global__ void interact_kernel(const float* __restrict__ qpart, const float* __restrict__ kpart,
                                const float* __restrict__ qimp,  const float* __restrict__ kimp,
                                const float* __restrict__ b_int1, const float* __restrict__ w_int2,
                                const float* __restrict__ b_int2, float* __restrict__ comb) {
    int idx = blockIdx.x * blockDim.x + threadIdx.x;
    int h = idx & 15;
    int j = (idx >> 4) & 63;
    int i = (idx >> 10) & 63;
    int b = idx >> 16;
    int qrow = (b * NB + i) * H_ + h;
    int krow = (b * NB + j) * H_ + h;
    const float* qp = qpart + (size_t)qrow * HID;
    const float* kp = kpart + (size_t)krow * HID;
    float s = b_int2[0];
    #pragma unroll
    for (int c = 0; c < HID; ++c) {
        float v = qp[c] + kp[c] + b_int1[c];
        s += fmaxf(v, 0.0f) * w_int2[c];
    }
    float inter = 1.0f / (1.0f + __expf(-s));
    comb[idx] = qimp[qrow] * kimp[krow] * inter;
}

// ---------------------------------------------------------------------------
// Kernel 5: per raw-reshape slice [b][hh][0..4095], select top 1024 by value
// bisection on the threshold (values in (0,1)), then write the boolean mask.
// The reference's scatter position equals the flat slice index, so output is
// simply mask[p] in order.
// ---------------------------------------------------------------------------
__global__ void topk_mask_kernel(const float* __restrict__ comb, float* __restrict__ out) {
    __shared__ float slo, shi;
    __shared__ int   scnt;
    int blk = blockIdx.x;                        // b*16 + hh  (64 slices)
    const float* base = comb + (size_t)blk * 4096;
    int tid = threadIdx.x;
    float local[16];
    #pragma unroll
    for (int t = 0; t < 16; ++t) local[t] = base[tid + t * 256];
    if (tid == 0) { slo = 0.0f; shi = 1.0f; }
    __syncthreads();
    for (int iter = 0; iter < 40; ++iter) {
        float mid = 0.5f * (slo + shi);
        int c = 0;
        #pragma unroll
        for (int t = 0; t < 16; ++t) c += (local[t] > mid) ? 1 : 0;
        if (tid == 0) scnt = 0;
        __syncthreads();
        atomicAdd(&scnt, c);
        __syncthreads();
        if (tid == 0) { if (scnt >= 1024) slo = mid; else shi = mid; }
        __syncthreads();
    }
    float thr = slo;
    #pragma unroll
    for (int t = 0; t < 16; ++t) {
        int p = tid + t * 256;
        out[(size_t)blk * 4096 + p] = (local[t] > thr) ? 1.0f : 0.0f;
    }
}

// ---------------------------------------------------------------------------
extern "C" void kernel_launch(void* const* d_in, const int* in_sizes, int n_in,
                              void* d_out, int out_size, void* d_ws, size_t ws_size,
                              hipStream_t stream) {
    const float* q      = (const float*)d_in[0];
    const float* k      = (const float*)d_in[1];
    const float* w_imp1 = (const float*)d_in[2];
    const float* b_imp1 = (const float*)d_in[3];
    const float* w_imp2 = (const float*)d_in[4];
    const float* b_imp2 = (const float*)d_in[5];
    const float* w_imp3 = (const float*)d_in[6];
    const float* b_imp3 = (const float*)d_in[7];
    const float* w_int1 = (const float*)d_in[8];
    const float* b_int1 = (const float*)d_in[9];
    const float* w_int2 = (const float*)d_in[10];
    const float* b_int2 = (const float*)d_in[11];
    float* out = (float*)d_out;

    char* w = (char*)d_ws;
    _Float16* qavg = (_Float16*)(w + (0u << 20));            // 1MB
    _Float16* kavg = (_Float16*)(w + (1u << 20));            // 1MB
    _Float16* w1h  = (_Float16*)(w + (2u << 20));            // 8KB
    _Float16* wi1h = (_Float16*)(w + (2u << 20) + 8192);     // 16KB
    float* qpart   = (float*)(w + (3u << 20));               // 512KB
    float* kpart   = (float*)(w + (3u << 20) + (512u << 10));// 512KB
    float* qimp    = (float*)(w + (4u << 20));               // 16KB
    float* kimp    = (float*)(w + (4u << 20) + (16u << 10)); // 16KB
    float* comb    = (float*)(w + (5u << 20));               // 1MB

    pool_kernel<<<1024, 256, 0, stream>>>(q, k, qavg, kavg);
    prep_w_kernel<<<32, 256, 0, stream>>>(w_imp1, w_int1, w1h, wi1h);
    mlp_wmma_kernel<<<512, 32, 0, stream>>>(qavg, kavg, w1h, wi1h,
                                            b_imp1, w_imp2, b_imp2, w_imp3, b_imp3,
                                            qpart, kpart, qimp, kimp);
    interact_kernel<<<1024, 256, 0, stream>>>(qpart, kpart, qimp, kimp,
                                              b_int1, w_int2, b_int2, comb);
    topk_mask_kernel<<<64, 256, 0, stream>>>(comb, out);
}
